// DNRI_DynamicVars_Encoder_52201032515963
// MI455X (gfx1250) — matrix operand order repaired and checked by reading, not compiled, not run on hardware
//
#include <hip/hip_runtime.h>
#include <cstddef>
#include <cstdint>

// Problem constants (from reference)
#define T_  64
#define N_  40
#define D_  4
#define H_  256
#define R_  64
#define E_  1560          // N*(N-1)
#define TE  (T_ * E_)     // 99840  (multiple of 16)
#define TN  (T_ * N_)     // 2560   (multiple of 16)
#define LSTM_BLKS ((E_ + 15) / 16)   // 98

typedef __attribute__((ext_vector_type(16))) __bf16 v16bf;
typedef __attribute__((ext_vector_type(8)))  __bf16 v8bf;
typedef __attribute__((ext_vector_type(8)))  float  v8f;

__device__ __forceinline__ float eluf(float v)  { return v > 0.f ? v : expm1f(v); }
__device__ __forceinline__ float sigmf(float v) { return 1.f / (1.f + expf(-v)); }

__device__ __forceinline__ v16bf cat16(v8bf lo, v8bf hi) {
  return __builtin_shufflevector(lo, hi, 0,1,2,3,4,5,6,7,8,9,10,11,12,13,14,15);
}

// ---------------------------------------------------------------------------
// fp32 -> bf16 weight conversion
// ---------------------------------------------------------------------------
__global__ void cvt_bf16(const float* __restrict__ s, __bf16* __restrict__ d, int n) {
  int i = blockIdx.x * 256 + threadIdx.x;
  if (i < n) d[i] = (__bf16)s[i];
}

// ---------------------------------------------------------------------------
// Small 2-layer ELU MLP, one row per block (used only for mlp1, Kin=4)
// ---------------------------------------------------------------------------
__global__ __launch_bounds__(256) void mlp_small(
    const float* __restrict__ X, int Kin,
    const float* __restrict__ W1, const float* __restrict__ b1,
    const float* __restrict__ W2, const float* __restrict__ b2,
    float* __restrict__ Y) {
  __shared__ float hb[H_];
  const int row = blockIdx.x, tid = threadIdx.x;
  const float* x = X + (size_t)row * Kin;
  float s = b1[tid];
  for (int k = 0; k < Kin; ++k) s += x[k] * W1[tid * Kin + k];
  hb[tid] = eluf(s);
  __syncthreads();
  float s2 = b2[tid];
  for (int k = 0; k < H_; ++k) s2 += hb[k] * W2[tid * H_ + k];
  Y[(size_t)row * H_ + tid] = eluf(s2);
}

// ---------------------------------------------------------------------------
// WMMA GEMM:  Y[M,256] = ELU(A[M,Kdim] @ W[256,Kdim]^T + bias)
//   MODE 0: A read directly from bf16 buffer Abf
//   MODE 1: A row (t*E+e) = concat(h[t,send[e],:], h[t,recv[e],:])   (Kdim=512)
//   MODE 2: A row = concat(n[t,send], n[t,recv], skip[t,e])          (Kdim=768)
//   MODE 3: A read directly from f32 buffer gsrc (cvt during stage)
// Block: 256 threads (8 waves) -> 16 rows x 256 cols; wave w owns cols 32w..32w+31
// ---------------------------------------------------------------------------
template <int MODE>
__global__ __launch_bounds__(256) void gemm16(
    const __bf16* __restrict__ Abf,
    const float*  __restrict__ gsrc,   // gather source [T,N,256] (or direct f32 A in MODE 3)
    const float*  __restrict__ gskip,  // [T,E,256] skip source
    const int*    __restrict__ send, const int* __restrict__ recv,
    const __bf16* __restrict__ W, const float* __restrict__ bias,
    int Kdim, float* __restrict__ outF, __bf16* __restrict__ outB) {
  __shared__ __bf16 As[16][32];
  const int tid  = threadIdx.x;
  const int lane = tid & 31;
  const int wave = tid >> 5;
  const int rowBase = blockIdx.x * 16;
  const int m_ = lane & 15;
  const int kb = (lane < 16) ? 0 : 8;    // unused; the A fragment is read at ko (k = 16h + s)
  const int ko = (lane < 16) ? 0 : 16;   // A and B fragment K base (contiguous k = 16h + s)
  const int n0 = wave * 32 + (lane & 15);
  const int n1 = n0 + 16;
  v8f acc0 = {}; v8f acc1 = {};
  const int nkc = Kdim >> 5;
  for (int kc = 0; kc < nkc; ++kc) {
    __syncthreads();                      // previous fragment reads done
    #pragma unroll
    for (int u = 0; u < 2; ++u) {         // stage 16x32 A tile as bf16
      int i = tid + 256 * u;              // 0..511
      int m = i >> 5, k = i & 31;
      int row = rowBase + m;
      int kg  = kc * 32 + k;
      float v;
      if (MODE == 0) {
        v = (float)Abf[(size_t)row * Kdim + kg];
      } else if (MODE == 1) {
        int t = row / E_, e = row - t * E_;
        int node = (kg < H_) ? send[e] : recv[e];
        v = gsrc[((size_t)t * N_ + node) * H_ + (kg & (H_ - 1))];
      } else if (MODE == 2) {
        int t = row / E_, e = row - t * E_;
        if (kg < H_)            v = gsrc[((size_t)t * N_ + send[e]) * H_ + kg];
        else if (kg < 2 * H_)   v = gsrc[((size_t)t * N_ + recv[e]) * H_ + (kg - H_)];
        else                    v = gskip[(size_t)row * H_ + (kg - 2 * H_)];
      } else {
        v = gsrc[(size_t)row * Kdim + kg];
      }
      As[m][k] = (__bf16)v;
    }
    __syncthreads();
    v8bf lo = *(const v8bf*)&As[m_][ko];
    v8bf hi = *(const v8bf*)&As[m_][ko + 8];
    v16bf a  = cat16(lo, hi);
    v16bf b0 = *(const v16bf*)(W + (size_t)n0 * Kdim + kc * 32 + ko);
    v16bf b1 = *(const v16bf*)(W + (size_t)n1 * Kdim + kc * 32 + ko);
    acc0 = __builtin_amdgcn_wmma_f32_16x16x32_bf16(false, a, false, b0, (short)0, acc0, false, false);
    acc1 = __builtin_amdgcn_wmma_f32_16x16x32_bf16(false, a, false, b1, (short)0, acc1, false, false);
  }
  const float bb0 = bias[n0], bb1 = bias[n1];
  #pragma unroll
  for (int j = 0; j < 8; ++j) {           // D layout: row = j + (lane<16?0:8)
    int row = rowBase + j + ((lane < 16) ? 0 : 8);
    float v0 = eluf(acc0[j] + bb0);
    float v1 = eluf(acc1[j] + bb1);
    size_t o = (size_t)row * H_;
    if (outF) { outF[o + n0] = v0;          outF[o + n1] = v1; }
    if (outB) { outB[o + n0] = (__bf16)v0;  outB[o + n1] = (__bf16)v1; }
  }
}

// ---------------------------------------------------------------------------
// Deterministic edge->node reduce (replaces scatter-add)
// ---------------------------------------------------------------------------
__global__ void build_lists(const int* __restrict__ recv, int* __restrict__ elist) {
  int r = threadIdx.x;
  if (r >= N_) return;
  int c = 0;
  for (int e = 0; e < E_; ++e)
    if (recv[e] == r) elist[r * (N_ - 1) + c++] = e;
}

__global__ __launch_bounds__(256) void edge2node(
    const float* __restrict__ e2, const int* __restrict__ elist,
    float* __restrict__ nbuf) {
  int tn = blockIdx.x, h = threadIdx.x;       // tn in [0, T*N)
  int t = tn / N_, r = tn - t * N_;
  float s = 0.f;
  for (int j = 0; j < N_ - 1; ++j) {
    int e = elist[r * (N_ - 1) + j];
    s += e2[((size_t)t * E_ + e) * H_ + h];
  }
  nbuf[(size_t)tn * H_ + h] = s;
}

// ---------------------------------------------------------------------------
// Persistent WMMA LSTM, both directions in one grid (2*LSTM_BLKS blocks):
// blocks [0,98) run forward, [98,196) run reverse -> overlapped recurrences.
// gates[16,256] = X_t[16,256] @ Wih^T + h[16,64] @ Whh^T + bih + bhh
// h,c,gates live in LDS; WMMA on bf16 X/weights, f32 state.
// ---------------------------------------------------------------------------
__global__ __launch_bounds__(256) void lstm_wmma2(
    const __bf16* __restrict__ X,                                    // [T,E,256] bf16
    const __bf16* __restrict__ fWih, const __bf16* __restrict__ fWhh,
    const float*  __restrict__ fbih, const float*  __restrict__ fbhh,
    float* __restrict__ fHout,                                       // [T,E,64]
    const __bf16* __restrict__ rWih, const __bf16* __restrict__ rWhh,
    const float*  __restrict__ rbih, const float*  __restrict__ rbhh,
    float* __restrict__ rHout) {
  const int reverse = (blockIdx.x >= LSTM_BLKS) ? 1 : 0;
  const __bf16* Wih  = reverse ? rWih  : fWih;
  const __bf16* Whh  = reverse ? rWhh  : fWhh;
  const float*  bih  = reverse ? rbih  : fbih;
  const float*  bhh  = reverse ? rbhh  : fbhh;
  float*        Hout = reverse ? rHout : fHout;
  __shared__ __bf16 Xs[16][H_];
  __shared__ float  hS[16][R_];
  __shared__ float  cS[16][R_];
  __shared__ float  gS[16][4 * R_];
  const int tid  = threadIdx.x;
  const int lane = tid & 31;
  const int wave = tid >> 5;
  const int rowBase = (blockIdx.x % LSTM_BLKS) * 16;
  const int m_ = lane & 15;
  const int kb = (lane < 16) ? 0 : 8;
  const int ko = (lane < 16) ? 0 : 16;
  const int n0 = wave * 32 + (lane & 15);
  const int n1 = n0 + 16;
  for (int i = tid; i < 16 * R_; i += 256) { (&hS[0][0])[i] = 0.f; (&cS[0][0])[i] = 0.f; }
  __syncthreads();
  const float bb0 = bih[n0] + bhh[n0], bb1 = bih[n1] + bhh[n1];
  for (int step = 0; step < T_; ++step) {
    int t = reverse ? (T_ - 1 - step) : step;
    for (int i = tid; i < 16 * H_; i += 256) {     // stage X_t tile
      int m = i >> 8, k = i & (H_ - 1);
      int row = rowBase + m;
      (&Xs[0][0])[i] = (row < E_) ? X[((size_t)t * E_ + row) * H_ + k] : (__bf16)0.f;
    }
    __syncthreads();
    v8f acc0 = {}; v8f acc1 = {};
    #pragma unroll
    for (int kc = 0; kc < H_ / 32; ++kc) {         // input contribution, K=256
      v16bf a  = cat16(*(const v8bf*)&Xs[m_][kc * 32 + ko],
                       *(const v8bf*)&Xs[m_][kc * 32 + ko + 8]);
      v16bf b0 = *(const v16bf*)(Wih + (size_t)n0 * H_ + kc * 32 + ko);
      v16bf b1 = *(const v16bf*)(Wih + (size_t)n1 * H_ + kc * 32 + ko);
      acc0 = __builtin_amdgcn_wmma_f32_16x16x32_bf16(false, a, false, b0, (short)0, acc0, false, false);
      acc1 = __builtin_amdgcn_wmma_f32_16x16x32_bf16(false, a, false, b1, (short)0, acc1, false, false);
    }
    #pragma unroll
    for (int kc = 0; kc < R_ / 32; ++kc) {         // recurrent contribution, K=64
      v16bf a;
      #pragma unroll
      for (int j = 0; j < 8; ++j) {
        a[j]     = (__bf16)hS[m_][kc * 32 + ko + j];
        a[8 + j] = (__bf16)hS[m_][kc * 32 + ko + 8 + j];
      }
      v16bf b0 = *(const v16bf*)(Whh + (size_t)n0 * R_ + kc * 32 + ko);
      v16bf b1 = *(const v16bf*)(Whh + (size_t)n1 * R_ + kc * 32 + ko);
      acc0 = __builtin_amdgcn_wmma_f32_16x16x32_bf16(false, a, false, b0, (short)0, acc0, false, false);
      acc1 = __builtin_amdgcn_wmma_f32_16x16x32_bf16(false, a, false, b1, (short)0, acc1, false, false);
    }
    #pragma unroll
    for (int j = 0; j < 8; ++j) {
      int mm = j + ((lane < 16) ? 0 : 8);
      gS[mm][n0] = acc0[j] + bb0;
      gS[mm][n1] = acc1[j] + bb1;
    }
    __syncthreads();
    for (int i = tid; i < 16 * R_; i += 256) {     // i,f,g,o gate order
      int mm = i >> 6, j = i & (R_ - 1);
      float ig = sigmf(gS[mm][j]);
      float fg = sigmf(gS[mm][R_ + j]);
      float gg = tanhf(gS[mm][2 * R_ + j]);
      float og = sigmf(gS[mm][3 * R_ + j]);
      float c = fg * cS[mm][j] + ig * gg;
      float h = og * tanhf(c);
      cS[mm][j] = c; hS[mm][j] = h;
      int row = rowBase + mm;
      if (row < E_) Hout[((size_t)t * E_ + row) * R_ + j] = h;
    }
    __syncthreads();
  }
}

// ---------------------------------------------------------------------------
// Final projection: out[t,e, 0:2]=prior(fwd_h), out[t,e, 2:4]=enc(concat fwd,rev)
// ---------------------------------------------------------------------------
__global__ __launch_bounds__(256) void project(
    const float* __restrict__ Fh, const float* __restrict__ Rh,
    const float* __restrict__ priW, const float* __restrict__ prib,
    const float* __restrict__ encW, const float* __restrict__ encb,
    float* __restrict__ out) {
  int idx = blockIdx.x * 256 + threadIdx.x;
  if (idx >= TE) return;
  const float* f = Fh + (size_t)idx * R_;
  const float* r = Rh + (size_t)idx * R_;
  #pragma unroll
  for (int k = 0; k < 2; ++k) {
    float p = prib[k], en = encb[k];
    for (int j = 0; j < R_; ++j) {
      p  += f[j] * priW[k * R_ + j];
      en += f[j] * encW[k * 2 * R_ + j] + r[j] * encW[k * 2 * R_ + R_ + j];
    }
    out[(size_t)idx * 4 + k]     = p;
    out[(size_t)idx * 4 + 2 + k] = en;
  }
}

// ---------------------------------------------------------------------------
extern "C" void kernel_launch(void* const* d_in, const int* in_sizes, int n_in,
                              void* d_out, int out_size, void* d_ws, size_t ws_size,
                              hipStream_t stream) {
  const float* x     = (const float*)d_in[0];
  const int*   send  = (const int*)d_in[2];
  const int*   recv  = (const int*)d_in[3];
  const float* m1W1 = (const float*)d_in[4],  *m1b1 = (const float*)d_in[5];
  const float* m1W2 = (const float*)d_in[6],  *m1b2 = (const float*)d_in[7];
  const float* m2W1 = (const float*)d_in[8],  *m2b1 = (const float*)d_in[9];
  const float* m2W2 = (const float*)d_in[10], *m2b2 = (const float*)d_in[11];
  const float* m3W1 = (const float*)d_in[12], *m3b1 = (const float*)d_in[13];
  const float* m3W2 = (const float*)d_in[14], *m3b2 = (const float*)d_in[15];
  const float* m4W1 = (const float*)d_in[16], *m4b1 = (const float*)d_in[17];
  const float* m4W2 = (const float*)d_in[18], *m4b2 = (const float*)d_in[19];
  const float* fWih = (const float*)d_in[20], *fWhh = (const float*)d_in[21];
  const float* fbih = (const float*)d_in[22], *fbhh = (const float*)d_in[23];
  const float* rWih = (const float*)d_in[24], *rWhh = (const float*)d_in[25];
  const float* rbih = (const float*)d_in[26], *rbhh = (const float*)d_in[27];
  const float* encW = (const float*)d_in[28], *encb = (const float*)d_in[29];
  const float* priW = (const float*)d_in[30], *prib = (const float*)d_in[31];
  float* out = (float*)d_out;

  char* w = (char*)d_ws;
  auto carve = [&](size_t bytes) -> char* {
    char* p = w; w += (bytes + 255) & ~(size_t)255; return p;
  };
  float*  h1    = (float*)carve((size_t)TN * H_ * 4);
  __bf16* e13   = (__bf16*)carve((size_t)TE * H_ * 2);   // mlp2-fc1 out, reused for mlp4-fc1 out
  float*  e2    = (float*)carve((size_t)TE * H_ * 4);    // mlp2 out == skip
  float*  nbuf  = (float*)carve((size_t)TN * H_ * 4);
  float*  n3    = (float*)carve((size_t)TN * H_ * 4);
  __bf16* h3b   = (__bf16*)carve((size_t)TN * H_ * 2);   // mlp3 hidden (bf16)
  __bf16* e4    = (__bf16*)carve((size_t)TE * H_ * 2);   // mlp4 out (LSTM input)
  float*  fh    = (float*)carve((size_t)TE * R_ * 4);
  float*  rh    = (float*)carve((size_t)TE * R_ * 4);
  __bf16* m2W1b = (__bf16*)carve((size_t)H_ * 2 * H_ * 2);
  __bf16* m2W2b = (__bf16*)carve((size_t)H_ * H_ * 2);
  __bf16* m3W1b = (__bf16*)carve((size_t)H_ * H_ * 2);
  __bf16* m3W2b = (__bf16*)carve((size_t)H_ * H_ * 2);
  __bf16* m4W1b = (__bf16*)carve((size_t)H_ * 3 * H_ * 2);
  __bf16* m4W2b = (__bf16*)carve((size_t)H_ * H_ * 2);
  __bf16* fWihb = (__bf16*)carve((size_t)4 * R_ * H_ * 2);
  __bf16* fWhhb = (__bf16*)carve((size_t)4 * R_ * R_ * 2);
  __bf16* rWihb = (__bf16*)carve((size_t)4 * R_ * H_ * 2);
  __bf16* rWhhb = (__bf16*)carve((size_t)4 * R_ * R_ * 2);
  int*    elist = (int*)carve((size_t)N_ * (N_ - 1) * 4);

  auto cvt = [&](const float* s, __bf16* d, int n) {
    cvt_bf16<<<(n + 255) / 256, 256, 0, stream>>>(s, d, n);
  };
  cvt(m2W1, m2W1b, H_ * 2 * H_);
  cvt(m2W2, m2W2b, H_ * H_);
  cvt(m3W1, m3W1b, H_ * H_);
  cvt(m3W2, m3W2b, H_ * H_);
  cvt(m4W1, m4W1b, H_ * 3 * H_);
  cvt(m4W2, m4W2b, H_ * H_);
  cvt(fWih, fWihb, 4 * R_ * H_);
  cvt(fWhh, fWhhb, 4 * R_ * R_);
  cvt(rWih, rWihb, 4 * R_ * H_);
  cvt(rWhh, rWhhb, 4 * R_ * R_);

  // 1) mlp1: [T*N, 4] -> h1 [T*N, 256]   (K=4, not WMMA material)
  mlp_small<<<TN, 256, 0, stream>>>(x, D_, m1W1, m1b1, m1W2, m1b2, h1);
  // 2) mlp2 fc1 with fused node2edge gather (K=512) -> e13 bf16
  gemm16<1><<<TE / 16, 256, 0, stream>>>(nullptr, h1, nullptr, send, recv,
                                         m2W1b, m2b1, 2 * H_, nullptr, e13);
  // 3) mlp2 fc2 (K=256) -> e2 f32 (= skip)
  gemm16<0><<<TE / 16, 256, 0, stream>>>(e13, nullptr, nullptr, nullptr, nullptr,
                                         m2W2b, m2b2, H_, e2, nullptr);
  // 4) edge2node reduce
  build_lists<<<1, 64, 0, stream>>>(recv, elist);
  edge2node<<<TN, 256, 0, stream>>>(e2, elist, nbuf);
  // 5) mlp3 via WMMA (fc1: f32 A direct; fc2: bf16 A)
  gemm16<3><<<TN / 16, 256, 0, stream>>>(nullptr, nbuf, nullptr, nullptr, nullptr,
                                         m3W1b, m3b1, H_, nullptr, h3b);
  gemm16<0><<<TN / 16, 256, 0, stream>>>(h3b, nullptr, nullptr, nullptr, nullptr,
                                         m3W2b, m3b2, H_, n3, nullptr);
  // 6) mlp4 fc1 with fused 3-way gather (K=768) -> e13 bf16
  gemm16<2><<<TE / 16, 256, 0, stream>>>(nullptr, n3, e2, send, recv,
                                         m4W1b, m4b1, 3 * H_, nullptr, e13);
  // 7) mlp4 fc2 -> e4 bf16 (LSTM input)
  gemm16<0><<<TE / 16, 256, 0, stream>>>(e13, nullptr, nullptr, nullptr, nullptr,
                                         m4W2b, m4b2, H_, nullptr, e4);
  // 8) forward + reverse LSTMs overlapped in a single grid
  lstm_wmma2<<<2 * LSTM_BLKS, 256, 0, stream>>>(e4,
                                                fWihb, fWhhb, fbih, fbhh, fh,
                                                rWihb, rWhhb, rbih, rbhh, rh);
  // 9) output projection -> [T, E, 4]
  project<<<(TE + 255) / 256, 256, 0, stream>>>(fh, rh, priW, prib, encW, encb, out);
}
